// EdgeRepresentationExtractor_9929964389089
// MI455X (gfx1250) — matrix operand order repaired and checked by reading, not compiled, not run on hardware
//
#include <hip/hip_runtime.h>
#include <hip/hip_bf16.h>

// N=4096, PAIR=2, M=64, D=128, O=128, R=2
// out[n,m,k] = sum_o h[n,0,m,o] * h[n,1,k,o]
// h = relu(deg * (X@Wsum + bsum) + (X@W0 + b0))

typedef __attribute__((ext_vector_type(16))) __bf16 v16bf;
typedef __attribute__((ext_vector_type(8)))  __bf16 v8bf;
typedef __attribute__((ext_vector_type(8)))  float  v8f;

union BF16x16 { v16bf v; v8bf h[2]; };

__device__ __forceinline__ unsigned short cvt_bf16(float f) {
  unsigned u = __float_as_uint(f);
  u += 0x7fffu + ((u >> 16) & 1u);          // round-to-nearest-even
  return (unsigned short)(u >> 16);
}

__device__ __forceinline__ v8bf ld8(const unsigned short* p) {
  return *reinterpret_cast<const v8bf*>(p);
}

// ---- prep: Wsum^T / W0^T as bf16, bsum as f32, into workspace ----
__global__ void prep_weights(const float* __restrict__ W_r,   // [2,128,128]
                             const float* __restrict__ b_r,   // [2,128]
                             const float* __restrict__ W0,    // [128,128]
                             unsigned short* __restrict__ WsumT, // [128(o),128(d)] bf16
                             unsigned short* __restrict__ W0T,   // [128(o),128(d)] bf16
                             float* __restrict__ bsum)           // [128]
{
  int i = blockIdx.x * blockDim.x + threadIdx.x;   // 0..16383
  if (i >= 128 * 128) return;
  int d = i >> 7, o = i & 127;
  float ws = W_r[d * 128 + o] + W_r[128 * 128 + d * 128 + o];
  WsumT[o * 128 + d] = cvt_bf16(ws);
  W0T[o * 128 + d]   = cvt_bf16(W0[d * 128 + o]);
  if (i < 128) bsum[i] = b_r[i] + b_r[128 + i];
}

// ---- main fused kernel: one block per node ----
// NOTE: WsumT and W0T must be CONTIGUOUS (W0T == WsumT + 128*128); the async
// weight staging copies 64 KB starting at WsumT straight into Ws[2][...].
__global__ __launch_bounds__(256) void edge_repr_main(
    const float* __restrict__ x,              // [N,2,64,128] f32
    const int*   __restrict__ deg,            // [N]
    const unsigned short* __restrict__ WsumT, // [O][D] bf16 (followed by W0T)
    const float* __restrict__ bsum,           // [O]
    const float* __restrict__ b0,             // [O]
    float* __restrict__ out)                  // [N,64,64]
{
  __shared__ alignas(16) unsigned short Xs[128 * 128];      // 32 KB, row = p*64+m
  __shared__ alignas(16) unsigned short Ws[2][128 * 128];   // 64 KB, [o][d]
  __shared__ alignas(16) unsigned short Hs[128 * 128];      // 32 KB, row = p*64+m, col = o

  const int n    = blockIdx.x;
  const int tid  = threadIdx.x;
  const int lane = tid & 31;
  const int wave = tid >> 5;
  const int g    = lane >> 4;      // half-wave select (0/1)
  const int ln   = lane & 15;

  // Stage A2 (issued first so it overlaps the VALU-heavy X conversion):
  // weights bf16 -> LDS via async DMA-style copy, tracked on ASYNCcnt.
  // 4096 x 16B = 64 KB covering Ws[0] then Ws[1].
  {
    const unsigned lds_base = (unsigned)(size_t)(&Ws[0][0]); // low 32b = LDS addr
    const unsigned long long g_base = (unsigned long long)(size_t)WsumT;
    for (int i = tid; i < 4096; i += 256) {
      unsigned laddr = lds_base + (unsigned)i * 16u;
      unsigned long long gaddr = g_base + (unsigned long long)i * 16ull;
      asm volatile("global_load_async_to_lds_b128 %0, %1, off"
                   :: "v"(laddr), "v"(gaddr) : "memory");
    }
  }

  // Stage A: X (64 KB f32) -> LDS bf16; 2 float4 -> 1 uint4 (ds_store_b128)
  {
    const float4* xg = reinterpret_cast<const float4*>(x + (size_t)n * (2 * 64 * 128));
    for (int i = tid; i < 2048; i += 256) {
      float4 a = xg[2 * i];
      float4 b = xg[2 * i + 1];
      uint4 pk;
      pk.x = (unsigned)cvt_bf16(a.x) | ((unsigned)cvt_bf16(a.y) << 16);
      pk.y = (unsigned)cvt_bf16(a.z) | ((unsigned)cvt_bf16(a.w) << 16);
      pk.z = (unsigned)cvt_bf16(b.x) | ((unsigned)cvt_bf16(b.y) << 16);
      pk.w = (unsigned)cvt_bf16(b.z) | ((unsigned)cvt_bf16(b.w) << 16);
      *reinterpret_cast<uint4*>(&Xs[i * 8]) = pk;
    }
  }

  asm volatile("s_wait_asynccnt 0x0" ::: "memory");
  __syncthreads();

  const float degf = (float)deg[n];

  // Stage B: per wave, 16-row strip: P = X@Wsum, Q = X@W0, h = relu(deg*(P+bs)+Q+b0)
  {
    const int Mb = wave * 16;
    for (int ct = 0; ct < 8; ++ct) {
      const int ob = ct * 16;
      v8f accp = {}; v8f accq = {};
#pragma unroll
      for (int k = 0; k < 4; ++k) {
        const int kb = k * 32;
        BF16x16 A, B0, B1;
        // A fragment: lane M=ln; K = kb + g*16 .. +15
        A.h[0] = ld8(&Xs[(Mb + ln) * 128 + kb + g * 16]);
        A.h[1] = ld8(&Xs[(Mb + ln) * 128 + kb + g * 16 + 8]);
        // B fragment: lane N=ln; K = kb + g*16 .. +15 (Ws is [o][d])
        B0.h[0] = ld8(&Ws[0][(ob + ln) * 128 + kb + g * 16]);
        B0.h[1] = ld8(&Ws[0][(ob + ln) * 128 + kb + g * 16 + 8]);
        B1.h[0] = ld8(&Ws[1][(ob + ln) * 128 + kb + g * 16]);
        B1.h[1] = ld8(&Ws[1][(ob + ln) * 128 + kb + g * 16 + 8]);
        accp = __builtin_amdgcn_wmma_f32_16x16x32_bf16(false, A.v, false, B0.v,
                                                       (short)0, accp, false, false);
        accq = __builtin_amdgcn_wmma_f32_16x16x32_bf16(false, A.v, false, B1.v,
                                                       (short)0, accq, false, false);
      }
      const int o = ob + ln;
      const float bs = bsum[o];
      const float bz = b0[o];
#pragma unroll
      for (int r = 0; r < 8; ++r) {
        float hv = degf * (accp[r] + bs) + (accq[r] + bz);
        hv = fmaxf(hv, 0.0f);
        // C layout: lane n-index = ln ; row M = r + 8*g
        Hs[(Mb + r + g * 8) * 128 + o] = cvt_bf16(hv);
      }
    }
  }
  __syncthreads();

  // Stage C: out[n] = h0 (rows 0..63) @ h1^T (rows 64..127); 16 tiles, 2 per wave
  {
    float* op = out + (size_t)n * 4096;
    for (int t = wave; t < 16; t += 8) {
      const int mt = (t >> 2) * 16;   // h0 row tile
      const int kt = (t & 3) * 16;    // h1 row tile == out col tile
      v8f acc = {};
#pragma unroll
      for (int k = 0; k < 4; ++k) {
        const int ob = k * 32;
        BF16x16 A, B;
        A.h[0] = ld8(&Hs[(mt + ln) * 128 + ob + g * 16]);
        A.h[1] = ld8(&Hs[(mt + ln) * 128 + ob + g * 16 + 8]);
        // B[o, j] = h1[j, o]: row-major Hs gives contiguous per-lane chunks
        B.h[0] = ld8(&Hs[(64 + kt + ln) * 128 + ob + g * 16]);
        B.h[1] = ld8(&Hs[(64 + kt + ln) * 128 + ob + g * 16 + 8]);
        acc = __builtin_amdgcn_wmma_f32_16x16x32_bf16(false, A.v, false, B.v,
                                                      (short)0, acc, false, false);
      }
#pragma unroll
      for (int r = 0; r < 8; ++r) {
        op[(mt + r + g * 8) * 64 + kt + ln] = acc[r];
      }
    }
  }
}

extern "C" void kernel_launch(void* const* d_in, const int* in_sizes, int n_in,
                              void* d_out, int out_size, void* d_ws, size_t ws_size,
                              hipStream_t stream) {
  const float* x    = (const float*)d_in[0];  // node_features [4096,2,64,128]
  const int*   deg  = (const int*)  d_in[1];  // [4096]
  const float* W_r  = (const float*)d_in[2];  // [2,128,128]
  const float* b_r  = (const float*)d_in[3];  // [2,128]
  const float* W0   = (const float*)d_in[4];  // [128,128]
  const float* b0   = (const float*)d_in[5];  // [128]
  float* out = (float*)d_out;                 // [4096,64,64]

  char* ws = (char*)d_ws;
  unsigned short* WsumT = (unsigned short*)ws;            // 32 KB
  unsigned short* W0T   = (unsigned short*)(ws + 32768);  // 32 KB (contiguous after WsumT)
  float*          bsum  = (float*)(ws + 65536);           // 512 B

  prep_weights<<<64, 256, 0, stream>>>(W_r, b_r, W0, WsumT, W0T, bsum);
  edge_repr_main<<<4096, 256, 0, stream>>>(x, deg, WsumT, bsum, b0, out);
}
